// TemporalEncoder_38354057953677
// MI455X (gfx1250) — matrix operand order repaired and checked by reading, not compiled, not run on hardware
//
#include <hip/hip_runtime.h>

typedef __attribute__((ext_vector_type(16))) _Float16 v16h;
typedef __attribute__((ext_vector_type(8)))  _Float16 v8h;
typedef __attribute__((ext_vector_type(8)))  float    v8f;
typedef __attribute__((ext_vector_type(4)))  unsigned int v4u;

#define AS3 __attribute__((address_space(3)))

#define HDIM    64
#define I_DIM   9
#define T_STEPS 30
#define B_TOT   65536
#define K0      96    // [x(9) pad(7) | h0(64) | pad(16)] zero-padded weights
#define K1      128   // [h0(64) | h1(64)]
#define NW      8     // waves per block

// generic -> LDS pointer (low 32 bits of a generic LDS address are the DS offset)
__device__ __forceinline__ AS3 const _Float16* lds_cp(const _Float16* p) {
  return (AS3 const _Float16*)(unsigned int)(unsigned long long)p;
}
__device__ __forceinline__ AS3 _Float16* lds_p(_Float16* p) {
  return (AS3 _Float16*)(unsigned int)(unsigned long long)p;
}

__device__ __forceinline__ v16h cat8(v8h lo, v8h hi) {
  return __builtin_shufflevector(lo, hi, 0,1,2,3,4,5,6,7,8,9,10,11,12,13,14,15);
}
// A 16x32 f16: lane m = lane%16, half kh = lane/16.
// element e <-> K = kbase + 16*kh + e (contiguous 16 f16)
__device__ __forceinline__ v16h load_a_frag(AS3 const _Float16* row, int kbase, int kh) {
  v8h lo = *(AS3 const v8h*)(row + kbase + 16 * kh);
  v8h hi = *(AS3 const v8h*)(row + kbase + 16 * kh + 8);
  return cat8(lo, hi);
}
// B 32x16 f16: lane n = lane%16; element e <-> K = koff + e (contiguous 16 f16).
// volatile + AS3: stays an in-loop ds_load_b128 pair (no LICM->scratch, no flat).
__device__ __forceinline__ v16h load_b_frag_v(AS3 const _Float16* wrow, int koff) {
  const volatile AS3 v4u* p = (const volatile AS3 v4u*)(wrow + koff);
  v4u lo = p[0];
  v4u hi = p[1];
  return cat8(__builtin_bit_cast(v8h, lo), __builtin_bit_cast(v8h, hi));
}
__device__ __forceinline__ float sigm(float x) { return 1.0f / (1.0f + __expf(-x)); }
__device__ __forceinline__ float tanh_f(float x) {
  x = fminf(fmaxf(x, -15.0f), 15.0f);
  const float e = __expf(2.0f * x);
  return (e - 1.0f) / (e + 1.0f);
}

// ---------- weight pack/transpose/pad into workspace ----------
__global__ void prep_kernel(const float* __restrict__ Wih0, const float* __restrict__ Whh0,
                            const float* __restrict__ bih0, const float* __restrict__ bhh0,
                            const float* __restrict__ Wih1, const float* __restrict__ Whh1,
                            const float* __restrict__ bih1, const float* __restrict__ bhh1,
                            _Float16* __restrict__ wc0, _Float16* __restrict__ wc1,
                            float* __restrict__ b0, float* __restrict__ b1) {
  const int n = blockIdx.x;          // 0..255 (gate-output index, PyTorch order i,f,g,o)
  const int t = threadIdx.x;         // 128
  for (int k = t; k < K0; k += 128) {
    float v = 0.0f;
    if (k < I_DIM)               v = Wih0[n * I_DIM + k];
    else if (k >= 16 && k < 80)  v = Whh0[n * HDIM + (k - 16)];
    wc0[n * K0 + k] = (_Float16)v;
  }
  for (int k = t; k < K1; k += 128) {
    const float v = (k < HDIM) ? Wih1[n * HDIM + k] : Whh1[n * HDIM + (k - HDIM)];
    wc1[n * K1 + k] = (_Float16)v;
  }
  if (t == 0) { b0[n] = bih0[n] + bhh0[n]; b1[n] = bih1[n] + bhh1[n]; }
}

// ---------- fused 2-layer LSTM + LayerNorm ----------
__global__ void __launch_bounds__(NW * 32)
lstm2_kernel(const float* __restrict__ x,
             const _Float16* __restrict__ wc0g, const _Float16* __restrict__ wc1g,
             const float* __restrict__ b0g, const float* __restrict__ b1g,
             const float* __restrict__ gamma, const float* __restrict__ beta,
             float* __restrict__ out) {
  __shared__ _Float16 wc0[256 * K0];            // 48 KB, [N][K] so B-frags are contiguous
  __shared__ _Float16 wc1[256 * K1];            // 64 KB
  __shared__ float    b0s[256];
  __shared__ float    b1s[256];
  __shared__ _Float16 hbuf[NW][16][144];        // per-wave [x|pad|h0|h1], 4.5 KB each

  const int tid  = threadIdx.x;
  const int wv   = tid >> 5;
  const int lane = tid & 31;
  const int lcol = lane & 15;
  const int kh   = lane >> 4;

  { // cooperative weight staging (b128)
    const uint4* s0p = (const uint4*)wc0g;  uint4* d0p = (uint4*)wc0;
    for (int i = tid; i < 256 * K0 * 2 / 16; i += NW * 32) d0p[i] = s0p[i];
    const uint4* s1p = (const uint4*)wc1g;  uint4* d1p = (uint4*)wc1;
    for (int i = tid; i < 256 * K1 * 2 / 16; i += NW * 32) d1p[i] = s1p[i];
    if (tid < 256) { b0s[tid] = b0g[tid]; b1s[tid] = b1g[tid]; }
  }
  { // zero this wave's h buffer (x-padding + initial h0/h1 state)
    _Float16* hb = &hbuf[wv][0][0];
    for (int i = lane; i < 16 * 144; i += 32) hb[i] = (_Float16)0.0f;
  }
  __syncthreads();

  const size_t s0 = ((size_t)blockIdx.x * NW + wv) * 16;   // first sample of this wave's tile

  AS3 const _Float16* w0l  = lds_cp(wc0);
  AS3 const _Float16* w1l  = lds_cp(wc1);
  AS3 const _Float16* arow = lds_cp(&hbuf[wv][lcol][0]);
  AS3 _Float16*       hrow = lds_p(&hbuf[wv][0][0]);       // per-wave base for h stores

  // hoist per-lane bias scalars out of the time loop (added in activation epilogue)
  float bv0[4][4], bv1[4][4];
  #pragma unroll
  for (int jt = 0; jt < 4; ++jt) {
    #pragma unroll
    for (int g = 0; g < 4; ++g) {
      bv0[jt][g] = b0s[g * 64 + jt * 16 + lcol];
      bv1[jt][g] = b1s[g * 64 + jt * 16 + lcol];
    }
  }

  v8f c0[4] = {}, c1[4] = {}, h1reg[4] = {};

  for (int t = 0; t < T_STEPS; ++t) {
    // stage x_t (16 samples x 9 features) into cols [0,9)
    for (int idx = lane; idx < 16 * I_DIM; idx += 32) {
      const int m = idx / I_DIM;
      const int k = idx - m * I_DIM;
      hrow[m * 144 + k] = (_Float16)x[(s0 + m) * (size_t)(T_STEPS * I_DIM) + t * I_DIM + k];
    }

    // ---- layer 0: gates = [x_t | h0] @ Wc0^T + b0  (K window [0,96)) ----
    {
      v16h a0[3];
      #pragma unroll
      for (int c = 0; c < 3; ++c) a0[c] = load_a_frag(arow, c * 32, kh);
      #pragma unroll
      for (int jt = 0; jt < 4; ++jt) {
        const int n = jt * 16 + lcol;
        v8f gi = {}, gf = {}, gg = {}, go = {};
        AS3 const _Float16* wi = w0l + n * K0;
        #pragma unroll
        for (int c = 0; c < 3; ++c) {
          const int ko = c * 32 + 16 * kh;
          v16h bfi = load_b_frag_v(wi, ko);
          gi = __builtin_amdgcn_wmma_f32_16x16x32_f16(false, a0[c], false, bfi, (short)0, gi, false, false);
          v16h bff = load_b_frag_v(wi,  64 * K0 + ko);
          gf = __builtin_amdgcn_wmma_f32_16x16x32_f16(false, a0[c], false, bff, (short)0, gf, false, false);
          v16h bfg = load_b_frag_v(wi, 128 * K0 + ko);
          gg = __builtin_amdgcn_wmma_f32_16x16x32_f16(false, a0[c], false, bfg, (short)0, gg, false, false);
          v16h bfo = load_b_frag_v(wi, 192 * K0 + ko);
          go = __builtin_amdgcn_wmma_f32_16x16x32_f16(false, a0[c], false, bfo, (short)0, go, false, false);
        }
        #pragma unroll
        for (int e = 0; e < 8; ++e) {
          const float iv = sigm(gi[e] + bv0[jt][0]);
          const float fv = sigm(gf[e] + bv0[jt][1]);
          const float gv = tanh_f(gg[e] + bv0[jt][2]);
          const float ov = sigm(go[e] + bv0[jt][3]);
          const float c  = fv * c0[jt][e] + iv * gv;
          c0[jt][e] = c;
          hrow[(e + 8 * kh) * 144 + 16 + jt * 16 + lcol] = (_Float16)(ov * tanh_f(c));
        }
      }
    }

    // ---- layer 1: gates = [h0 | h1] @ Wc1^T + b1  (K window [16,144)) ----
    {
      v16h a1[4];
      #pragma unroll
      for (int c = 0; c < 4; ++c) a1[c] = load_a_frag(arow, 16 + c * 32, kh);
      #pragma unroll
      for (int jt = 0; jt < 4; ++jt) {
        const int n = jt * 16 + lcol;
        v8f gi = {}, gf = {}, gg = {}, go = {};
        AS3 const _Float16* wi = w1l + n * K1;
        #pragma unroll
        for (int c = 0; c < 4; ++c) {
          const int ko = c * 32 + 16 * kh;
          v16h bfi = load_b_frag_v(wi, ko);
          gi = __builtin_amdgcn_wmma_f32_16x16x32_f16(false, a1[c], false, bfi, (short)0, gi, false, false);
          v16h bff = load_b_frag_v(wi,  64 * K1 + ko);
          gf = __builtin_amdgcn_wmma_f32_16x16x32_f16(false, a1[c], false, bff, (short)0, gf, false, false);
          v16h bfg = load_b_frag_v(wi, 128 * K1 + ko);
          gg = __builtin_amdgcn_wmma_f32_16x16x32_f16(false, a1[c], false, bfg, (short)0, gg, false, false);
          v16h bfo = load_b_frag_v(wi, 192 * K1 + ko);
          go = __builtin_amdgcn_wmma_f32_16x16x32_f16(false, a1[c], false, bfo, (short)0, go, false, false);
        }
        #pragma unroll
        for (int e = 0; e < 8; ++e) {
          const float iv = sigm(gi[e] + bv1[jt][0]);
          const float fv = sigm(gf[e] + bv1[jt][1]);
          const float gv = tanh_f(gg[e] + bv1[jt][2]);
          const float ov = sigm(go[e] + bv1[jt][3]);
          const float c  = fv * c1[jt][e] + iv * gv;
          c1[jt][e] = c;
          const float h  = ov * tanh_f(c);
          h1reg[jt][e] = h;
          hrow[(e + 8 * kh) * 144 + 80 + jt * 16 + lcol] = (_Float16)h;
        }
      }
    }
  }

  // ---- LayerNorm over final h1, f32 from registers (butterfly over 16 lanes) ----
  float sum[8], sq[8];
  #pragma unroll
  for (int e = 0; e < 8; ++e) {
    float a = 0.0f, b = 0.0f;
    #pragma unroll
    for (int jt = 0; jt < 4; ++jt) { const float h = h1reg[jt][e]; a += h; b += h * h; }
    sum[e] = a; sq[e] = b;
  }
  #pragma unroll
  for (int mask = 1; mask < 16; mask <<= 1) {
    #pragma unroll
    for (int e = 0; e < 8; ++e) {
      sum[e] += __shfl_xor(sum[e], mask, 32);
      sq[e]  += __shfl_xor(sq[e],  mask, 32);
    }
  }
  float gm[4], bt[4];
  #pragma unroll
  for (int jt = 0; jt < 4; ++jt) {
    const int j = jt * 16 + lcol;
    gm[jt] = gamma[j]; bt[jt] = beta[j];
  }
  #pragma unroll
  for (int e = 0; e < 8; ++e) {
    const float mu  = sum[e] * (1.0f / 64.0f);
    const float var = sq[e] * (1.0f / 64.0f) - mu * mu;
    const float rs  = rsqrtf(var + 1e-5f);
    const size_t m  = s0 + e + 8 * kh;
    #pragma unroll
    for (int jt = 0; jt < 4; ++jt) {
      out[m * 64 + jt * 16 + lcol] = (h1reg[jt][e] - mu) * rs * gm[jt] + bt[jt];
    }
  }
}

extern "C" void kernel_launch(void* const* d_in, const int* in_sizes, int n_in,
                              void* d_out, int out_size, void* d_ws, size_t ws_size,
                              hipStream_t stream) {
  (void)in_sizes; (void)n_in; (void)out_size; (void)ws_size;
  const float* x     = (const float*)d_in[0];
  const float* Wih0  = (const float*)d_in[1];
  const float* Whh0  = (const float*)d_in[2];
  const float* bih0  = (const float*)d_in[3];
  const float* bhh0  = (const float*)d_in[4];
  const float* Wih1  = (const float*)d_in[5];
  const float* Whh1  = (const float*)d_in[6];
  const float* bih1  = (const float*)d_in[7];
  const float* bhh1  = (const float*)d_in[8];
  const float* gamma = (const float*)d_in[9];
  const float* beta  = (const float*)d_in[10];

  _Float16* wc0 = (_Float16*)d_ws;              // 256*96  f16 = 48 KB
  _Float16* wc1 = wc0 + 256 * K0;               // 256*128 f16 = 64 KB
  float*    b0  = (float*)(wc1 + 256 * K1);     // 1 KB
  float*    b1  = b0 + 256;                     // 1 KB

  prep_kernel<<<256, 128, 0, stream>>>(Wih0, Whh0, bih0, bhh0,
                                       Wih1, Whh1, bih1, bhh1, wc0, wc1, b0, b1);
  lstm2_kernel<<<B_TOT / (NW * 16), NW * 32, 0, stream>>>(
      x, wc0, wc1, b0, b1, gamma, beta, (float*)d_out);
}
